// MultiHeadAttentionLayer_85615878079179
// MI455X (gfx1250) — matrix operand order repaired and checked by reading, not compiled, not run on hardware
//
#include <hip/hip_runtime.h>
#include <stdint.h>

typedef __attribute__((ext_vector_type(16))) _Float16 v16h;
typedef __attribute__((ext_vector_type(8)))  float    v8f;
typedef __attribute__((ext_vector_type(4)))  float    f32x4;

#define DIM   128
#define SEQ   4096
#define BATCH 4
#define KV    32      // keys per tile
#define WAVES 8
#define KSTR  136     // K tile LDS stride (halves): 272B -> bank start 4n, conflict-free
#define VSTR  40      // Vt tile LDS stride (halves): 80B  -> bank start 20n mod 64, conflict-free
#define PSTR  40      // P scratch stride (halves)

// ---------------- QKV projection: x[row,:] @ {Wq,Wk,Wv} -> f16 (V transposed) --------
__global__ __launch_bounds__(DIM)
void qkv_project(const float* __restrict__ x,
                 const float* __restrict__ wq,
                 const float* __restrict__ wk,
                 const float* __restrict__ wv,
                 _Float16* __restrict__ q,
                 _Float16* __restrict__ k,
                 _Float16* __restrict__ vt) {       // vt: [B][DIM][SEQ]
  __shared__ float xrow[DIM];
  const int row = blockIdx.x;            // b*SEQ + s
  const int e   = threadIdx.x;           // output column
  xrow[e] = x[(size_t)row * DIM + e];
  __syncthreads();
  float aq = 0.f, ak = 0.f, av = 0.f;
#pragma unroll 8
  for (int d = 0; d < DIM; ++d) {
    const float xv = xrow[d];
    aq = fmaf(xv, wq[d * DIM + e], aq);
    ak = fmaf(xv, wk[d * DIM + e], ak);
    av = fmaf(xv, wv[d * DIM + e], av);
  }
  const size_t o = (size_t)row * DIM + e;
  q[o] = (_Float16)aq;
  k[o] = (_Float16)ak;
  const int b = row >> 12, s = row & (SEQ - 1);
  vt[((size_t)b * DIM + e) * SEQ + s] = (_Float16)av;
}

__global__ void zero_out(float* __restrict__ out, int n) {
  int i = blockIdx.x * blockDim.x + threadIdx.x;
  if (i < n) out[i] = 0.f;
}

// ---- DPP16 butterfly reductions over a 16-lane half (no LDS traffic) ----
template <int CTRL>
__device__ __forceinline__ float dpp_f(float x) {
  return __int_as_float(
      __builtin_amdgcn_update_dpp(0, __float_as_int(x), CTRL, 0xf, 0xf, true));
}
__device__ __forceinline__ float rowmax16(float x) {
  x = fmaxf(x, dpp_f<0xB1>(x));   // quad_perm [1,0,3,2]  : xor 1
  x = fmaxf(x, dpp_f<0x4E>(x));   // quad_perm [2,3,0,1]  : xor 2
  x = fmaxf(x, dpp_f<0x141>(x));  // row_half_mirror      : completes group of 8
  x = fmaxf(x, dpp_f<0x140>(x));  // row_mirror           : completes group of 16
  return x;
}
__device__ __forceinline__ float rowsum16(float x) {
  x += dpp_f<0xB1>(x);
  x += dpp_f<0x4E>(x);
  x += dpp_f<0x141>(x);
  x += dpp_f<0x140>(x);
  return x;
}

// async global -> LDS copy, 16B per call, tracked by ASYNCcnt
__device__ __forceinline__ void async_b128(uint32_t lds_off, const void* gaddr) {
  asm volatile("global_load_async_to_lds_b128 %0, %1, off"
               :: "v"(lds_off), "v"(gaddr) : "memory");
}
__device__ __forceinline__ void wait_async0() {
  asm volatile("s_wait_asynccnt 0x0" ::: "memory");
}

// -------- fused flash attention + sum over queries -> out[b, d] --------
__global__ __launch_bounds__(256)
void flash_attn_sum(const _Float16* __restrict__ Q,
                    const _Float16* __restrict__ K,
                    const _Float16* __restrict__ Vt,   // [B][DIM][SEQ]
                    float* __restrict__ out) {
  __shared__ __align__(16) _Float16 klds[2][KV * KSTR];    // 2 x 8.5 KB [key][d] padded
  __shared__ __align__(16) _Float16 vtlds[2][DIM * VSTR];  // 2 x 10 KB  [d][key] padded
  __shared__ __align__(16) _Float16 pl[WAVES][16 * PSTR];  // 10 KB per-wave P scratch

  const int tid  = threadIdx.x;
  const int wave = tid >> 5;
  const int lane = tid & 31;
  const int n16  = lane & 15;   // B-layout / C-layout column
  const int h    = lane >> 4;   // lane half
  const int b    = blockIdx.y;
  const int qbase = blockIdx.x * (WAVES * 16) + wave * 16;

  // staging geometry (per thread, per tile): K: 2x16B of row [key][d0..]; Vt: 2x16B of row [d][k..]
  const int skey = tid >> 3;            // 0..31
  const int sd0  = (tid & 7) * 16;      // 0..112
  const int svd  = tid >> 1;            // 0..127 (d row of Vt)
  const int svk  = (tid & 1) * 16;      // key offset 0/16

  // ---- load Q tile in WMMA A-layout (16x32 f16 per K-chunk), reused all iterations
  v16h a_q[4];
  {
    const _Float16* qrow = Q + ((size_t)(b * SEQ) + qbase + n16) * DIM;
#pragma unroll
    for (int kk = 0; kk < 4; ++kk) {
      ((f32x4*)&a_q[kk])[0] = *(const f32x4*)(qrow + kk * 32 + h * 16);      // e=0..7  -> K=h*16+e
      ((f32x4*)&a_q[kk])[1] = *(const f32x4*)(qrow + kk * 32 + h * 16 + 8); // e=8..15 -> K=h*16+e
    }
  }

  float rowmax[8], rowsum[8];
  v8f acc[8];
#pragma unroll
  for (int e = 0; e < 8; ++e) { rowmax[e] = -3.0e38f; rowsum[e] = 0.f; }
#pragma unroll
  for (int dd = 0; dd < 8; ++dd) acc[dd] = (v8f){};

  const float scale = 0.08838834764831845f; // 1/sqrt(128)

  // ---- async stage of one KV tile into LDS buffer `nb`
  auto stage = [&](int nb, int kbase) {
    const _Float16* kg = K + ((size_t)(b * SEQ) + kbase + skey) * DIM + sd0;
    uint32_t ko = (uint32_t)(uintptr_t)&klds[nb][skey * KSTR + sd0];
    async_b128(ko,      kg);
    async_b128(ko + 16, kg + 8);
    const _Float16* vg = Vt + ((size_t)(b * DIM) + svd) * SEQ + kbase + svk;
    uint32_t vo = (uint32_t)(uintptr_t)&vtlds[nb][svd * VSTR + svk];
    async_b128(vo,      vg);
    async_b128(vo + 16, vg + 8);
  };

  stage(0, 0);
  wait_async0();
  __syncthreads();

  for (int kt = 0; kt < SEQ / KV; ++kt) {
    const int cur = kt & 1;
    if (kt + 1 < SEQ / KV) stage(cur ^ 1, (kt + 1) * KV);   // prefetch next tile (overlaps compute)

    const _Float16* kb = &klds[cur][0];
    const _Float16* vb = &vtlds[cur][0];

    // ---- scores S = Q Kt^T : two 16x16 f32 tiles (keys 0..15 / 16..31)
    v8f s0 = (v8f){}, s1 = (v8f){};
#pragma unroll
    for (int kk = 0; kk < 4; ++kk) {
      v16h bk0, bk1;
      const _Float16* p0 = kb + n16 * KSTR + kk * 32 + h * 16;        // B elem e -> K=h*16+e
      const _Float16* p1 = kb + (n16 + 16) * KSTR + kk * 32 + h * 16;
      ((f32x4*)&bk0)[0] = *(const f32x4*)p0;
      ((f32x4*)&bk0)[1] = *(const f32x4*)(p0 + 8);
      ((f32x4*)&bk1)[0] = *(const f32x4*)p1;
      ((f32x4*)&bk1)[1] = *(const f32x4*)(p1 + 8);
      s0 = __builtin_amdgcn_wmma_f32_16x16x32_f16(false, a_q[kk], false, bk0,
                                                  (short)0, s0, false, false);
      s1 = __builtin_amdgcn_wmma_f32_16x16x32_f16(false, a_q[kk], false, bk1,
                                                  (short)0, s1, true, false);  // reuse A
    }

    // ---- online softmax (DPP reductions); P -> per-wave LDS scratch, f16 [m][n]
    float corr[8];
#pragma unroll
    for (int e = 0; e < 8; ++e) {
      const float x0 = s0[e] * scale;
      const float x1 = s1[e] * scale;
      const float mloc = rowmax16(fmaxf(x0, x1));
      const float mnew = fmaxf(rowmax[e], mloc);
      corr[e] = __expf(rowmax[e] - mnew);
      rowmax[e] = mnew;
      const float p0 = __expf(x0 - mnew);
      const float p1 = __expf(x1 - mnew);
      rowsum[e] = rowsum[e] * corr[e] + rowsum16(p0 + p1);
      _Float16* pp = &pl[wave][(e + h * 8) * PSTR];
      pp[n16]      = (_Float16)p0;
      pp[n16 + 16] = (_Float16)p1;
    }
#pragma unroll
    for (int dd = 0; dd < 8; ++dd)
#pragma unroll
      for (int e = 0; e < 8; ++e) acc[dd][e] *= corr[e];

    // ---- re-read P in A-layout (16x32, K = key-in-tile)
    v16h a_p;
    {
      const _Float16* pp = &pl[wave][n16 * PSTR];
      ((f32x4*)&a_p)[0] = *(const f32x4*)(pp + h * 16);
      ((f32x4*)&a_p)[1] = *(const f32x4*)(pp + h * 16 + 8);
    }

    // ---- ctx += P x V : 8 WMMAs over d-chunks, V^T tile read contiguously
    // dd = 0 (no reuse flag), then dd = 1..7 with reuse-A hint (identical instr chain)
    {
      v16h bv;
      const _Float16* p = vb + n16 * VSTR + h * 16;
      ((f32x4*)&bv)[0] = *(const f32x4*)p;
      ((f32x4*)&bv)[1] = *(const f32x4*)(p + 8);
      acc[0] = __builtin_amdgcn_wmma_f32_16x16x32_f16(false, a_p, false, bv,
                                                      (short)0, acc[0], false, false);
    }
#pragma unroll
    for (int dd = 1; dd < 8; ++dd) {
      v16h bv;
      const _Float16* p = vb + (dd * 16 + n16) * VSTR + h * 16;   // B elem e -> key=h*16+e
      ((f32x4*)&bv)[0] = *(const f32x4*)p;
      ((f32x4*)&bv)[1] = *(const f32x4*)(p + 8);
      acc[dd] = __builtin_amdgcn_wmma_f32_16x16x32_f16(false, a_p, false, bv,
                                                       (short)0, acc[dd], true, false);
    }

    wait_async0();      // next tile's async DMAs have landed
    __syncthreads();    // everyone done reading `cur`, next buffer valid for all
  }

  // ---- normalize rows, sum the wave's 16 query rows, atomically add into out[b,:]
  float inv[8];
#pragma unroll
  for (int e = 0; e < 8; ++e) inv[e] = 1.0f / rowsum[e];
#pragma unroll
  for (int dd = 0; dd < 8; ++dd) {
    float cs = 0.f;
#pragma unroll
    for (int e = 0; e < 8; ++e) cs += acc[dd][e] * inv[e];
    cs += __shfl_xor(cs, 16, 32);   // combine m=0..7 / m=8..15 halves (same d)
    if (lane < 16) atomicAdd(&out[b * DIM + dd * 16 + n16], cs);
  }
}

extern "C" void kernel_launch(void* const* d_in, const int* in_sizes, int n_in,
                              void* d_out, int out_size, void* d_ws, size_t ws_size,
                              hipStream_t stream) {
  const float* x  = (const float*)d_in[0];
  const float* wq = (const float*)d_in[1];
  const float* wk = (const float*)d_in[2];
  const float* wv = (const float*)d_in[3];
  float* out = (float*)d_out;

  const size_t nqkv = (size_t)BATCH * SEQ * DIM;   // 2,097,152 elements
  _Float16* q  = (_Float16*)d_ws;
  _Float16* k  = q + nqkv;
  _Float16* vt = k + nqkv;                         // [B][DIM][SEQ]

  zero_out<<<2, 256, 0, stream>>>(out, BATCH * DIM);
  qkv_project<<<BATCH * SEQ, DIM, 0, stream>>>(x, wq, wk, wv, q, k, vt);
  dim3 grid(SEQ / (WAVES * 16), BATCH);            // 32 x 4 blocks, 256 threads
  flash_attn_sum<<<grid, 256, 0, stream>>>(q, k, vt, out);
}
